// PMLE_12850542150063
// MI455X (gfx1250) — matrix operand order repaired and checked by reading, not compiled, not run on hardware
//
#include <hip/hip_runtime.h>

// ---------------------------------------------------------------------------
// Types for CDNA5 WMMA (wave32)
// ---------------------------------------------------------------------------
typedef __attribute__((ext_vector_type(16))) __bf16 v16bf;
typedef __attribute__((ext_vector_type(8)))  float  v8f;

union FragBF { v16bf v; uint4 q[2]; };
union Pack4  { __bf16 h[4]; uint2 u; };
union Pack8  { __bf16 h[8]; uint4 u; };

#define EPS_BN 1e-5f

// ---------------------------------------------------------------------------
// Small helper kernels
// ---------------------------------------------------------------------------
__global__ void zero_f32_kernel(float* __restrict__ p, int n) {
    int i = blockIdx.x * blockDim.x + threadIdx.x;
    if (i < n) p[i] = 0.0f;
}

// Column-wise partial sums over a row-chunk; atomically accumulated.
__global__ void colsum_f32_kernel(const float* __restrict__ x,
                                  float* __restrict__ sums,
                                  float* __restrict__ sumsq,
                                  int N, int F, int rpb) {
    int f = blockIdx.x * blockDim.x + threadIdx.x;
    if (f >= F) return;
    int r0 = blockIdx.y * rpb;
    int r1 = r0 + rpb; if (r1 > N) r1 = N;
    float s = 0.0f, ss = 0.0f;
    for (int n = r0; n < r1; ++n) {
        float v = x[(size_t)n * F + f];
        s += v; ss += v * v;
    }
    atomicAdd(&sums[f], s);
    atomicAdd(&sumsq[f], ss);
}

__global__ void colsum_bf16_kernel(const void* __restrict__ xv,
                                   float* __restrict__ sums,
                                   float* __restrict__ sumsq,
                                   int N, int F, int rpb) {
    const __bf16* x = (const __bf16*)xv;
    int f = blockIdx.x * blockDim.x + threadIdx.x;
    if (f >= F) return;
    int r0 = blockIdx.y * rpb;
    int r1 = r0 + rpb; if (r1 > N) r1 = N;
    float s = 0.0f, ss = 0.0f;
    for (int n = r0; n < r1; ++n) {
        float v = (float)x[(size_t)n * F + f];
        s += v; ss += v * v;
    }
    atomicAdd(&sums[f], s);
    atomicAdd(&sumsq[f], ss);
}

// scale[f] = gamma[f] * rsqrt(var + eps); shift[f] = beta[f] - mean*scale
__global__ void bn_finalize_kernel(const float* __restrict__ sums,
                                   const float* __restrict__ sumsq,
                                   const float* __restrict__ g,
                                   const float* __restrict__ b,
                                   float* __restrict__ scale,
                                   float* __restrict__ shift,
                                   int N, int F) {
    int f = blockIdx.x * blockDim.x + threadIdx.x;
    if (f >= F) return;
    float invN  = 1.0f / (float)N;
    float mean  = sums[f] * invN;
    float var   = sumsq[f] * invN - mean * mean;
    float sc    = g[f] * rsqrtf(var + EPS_BN);
    scale[f] = sc;
    shift[f] = b[f] - mean * sc;
}

// Wb[row, col] = bf16( W[row, col] * scale[col] )   (K-dim scale fold)
__global__ void scale_w_kernel(const float* __restrict__ W,
                               const float* __restrict__ scale,
                               void* __restrict__ Wbv,
                               int K) {
    __bf16* Wb = (__bf16*)Wbv;
    int col = (blockIdx.x * blockDim.x + threadIdx.x) * 4;
    int row = blockIdx.y;
    if (col >= K) return;
    size_t base = (size_t)row * K + col;
    float4 wv = *reinterpret_cast<const float4*>(W + base);
    float4 sv = *reinterpret_cast<const float4*>(scale + col);
    Pack4 o;
    o.h[0] = (__bf16)(wv.x * sv.x);
    o.h[1] = (__bf16)(wv.y * sv.y);
    o.h[2] = (__bf16)(wv.z * sv.z);
    o.h[3] = (__bf16)(wv.w * sv.w);
    *reinterpret_cast<uint2*>(Wb + base) = o.u;
}

// bout[j] = bin[j] + sum_k shift[k] * W[j,k]
__global__ __launch_bounds__(256) void fold_bias_kernel(const float* __restrict__ W,
                                                        const float* __restrict__ shift,
                                                        const float* __restrict__ bin,
                                                        float* __restrict__ bout,
                                                        int K) {
    __shared__ float red[256];
    const float* w = W + (size_t)blockIdx.x * K;
    float s = 0.0f;
    for (int k = threadIdx.x; k < K; k += 256) s += shift[k] * w[k];
    red[threadIdx.x] = s;
    __syncthreads();
    for (int off = 128; off > 0; off >>= 1) {
        if (threadIdx.x < off) red[threadIdx.x] += red[threadIdx.x + off];
        __syncthreads();
    }
    if (threadIdx.x == 0) bout[blockIdx.x] = bin[blockIdx.x] + red[0];
}

// ---------------------------------------------------------------------------
// Async global->LDS helpers (CDNA5 GLOBAL_LOAD_ASYNC_TO_LDS_B128, ASYNCcnt)
// ---------------------------------------------------------------------------
__device__ __forceinline__ void async_copy_b128(unsigned lds_addr,
                                                unsigned long long gbase,
                                                unsigned goff_bytes) {
    // GVS mode: mem = SGPR_U64 + VGPR_I32 ; per lane copies 16B into LDS[lds_addr]
    asm volatile("global_load_async_to_lds_b128 %0, %1, %2"
                 :
                 : "v"(lds_addr), "v"(goff_bytes), "s"(gbase)
                 : "memory");
}

__device__ __forceinline__ void wait_async0() {
    asm volatile("s_wait_asynccnt 0x0" ::: "memory");
}

// ---------------------------------------------------------------------------
// WMMA GEMM:  C[M,N] = A[M,K] * B[N,K]^T  (+ epilogue)
//   ATYPE: 0 = fp32 A (register-staged, converted to bf16), 1 = bf16 A (async)
//   EPI:   0 = relu(acc + bias) -> bf16 C
//          1 = acc + bias + beta*Y -> fp32 C
// Block: 256 threads = 8 wave32s arranged 2(M) x 4(N); tile 128x128, BK=64.
// Double-buffered LDS; B (and bf16 A) staged with GLOBAL_LOAD_ASYNC_TO_LDS.
// ---------------------------------------------------------------------------
template<int ATYPE, int EPI>
__global__ __launch_bounds__(256) void wmma_gemm_kernel(
    const void* __restrict__ Av, const void* __restrict__ Bv,
    const float* __restrict__ bias, const float* __restrict__ Y,
    const float* __restrict__ betaPtr, void* __restrict__ Cout,
    int M, int N, int K)
{
    constexpr int BM = 128, BN = 128, BK = 64;
    constexpr int LDT = BK + 8;                 // 72 bf16 elems, rows 16B aligned
    __shared__ __bf16 sA[2][BM * LDT];
    __shared__ __bf16 sB[2][BN * LDT];

    const unsigned long long Abase = (unsigned long long)Av;
    const unsigned long long Bbase = (unsigned long long)Bv;

    const int bm = blockIdx.y * BM;
    const int bn = blockIdx.x * BN;
    const int t  = threadIdx.x;

    const int lane   = t & 31;
    const int wave   = t >> 5;          // 0..7
    const int waveM  = wave & 1;        // 2 wave-rows
    const int waveN  = wave >> 1;       // 4 wave-cols
    const int lane16 = lane & 15;
    const int laneHi = lane >> 4;

    v8f acc[4][2];
#pragma unroll
    for (int i = 0; i < 4; ++i)
#pragma unroll
        for (int j = 0; j < 2; ++j)
#pragma unroll
            for (int e = 0; e < 8; ++e) acc[i][j][e] = 0.0f;

    // Staging map: each thread moves 4 rows x 8 elems (8 threads per 64-wide row)
    const int sRow = t >> 3;            // 0..31 (row within 32-row pass)
    const int sCol = (t & 7) * 8;       // 0..56

    float4 raf[4][2];                   // fp32 A staging registers (ATYPE==0)

    // Issue async copies for one K-stage into buffer `buf`.
    auto issue_async = [&](int kt, int buf) {
#pragma unroll
        for (int p = 0; p < 4; ++p) {
            int r = p * 32 + sRow;
            unsigned ldsB =
                (unsigned)(uintptr_t)(&sB[buf][r * LDT + sCol]);
            unsigned goffB = (unsigned)(((size_t)(bn + r) * K + kt + sCol) * 2);
            async_copy_b128(ldsB, Bbase, goffB);
            if constexpr (ATYPE == 1) {
                unsigned ldsA =
                    (unsigned)(uintptr_t)(&sA[buf][r * LDT + sCol]);
                unsigned goffA = (unsigned)(((size_t)(bm + r) * K + kt + sCol) * 2);
                async_copy_b128(ldsA, Abase, goffA);
            }
        }
    };

    // fp32 A path: global -> registers
    auto load_regsA = [&](int kt) {
        if constexpr (ATYPE == 0) {
            const float* A = (const float*)Av;
#pragma unroll
            for (int p = 0; p < 4; ++p) {
                int r = p * 32 + sRow;
                const float4* g =
                    reinterpret_cast<const float4*>(A + (size_t)(bm + r) * K + kt + sCol);
                raf[p][0] = g[0];
                raf[p][1] = g[1];
            }
            if (kt + BK < K)
                __builtin_prefetch(A + (size_t)(bm + sRow) * K + kt + BK + sCol, 0, 0);
        }
    };

    // fp32 A path: registers -> LDS (convert to bf16)
    auto store_regsA = [&](int buf) {
        if constexpr (ATYPE == 0) {
#pragma unroll
            for (int p = 0; p < 4; ++p) {
                int r = p * 32 + sRow;
                Pack8 cv;
                cv.h[0] = (__bf16)raf[p][0].x; cv.h[1] = (__bf16)raf[p][0].y;
                cv.h[2] = (__bf16)raf[p][0].z; cv.h[3] = (__bf16)raf[p][0].w;
                cv.h[4] = (__bf16)raf[p][1].x; cv.h[5] = (__bf16)raf[p][1].y;
                cv.h[6] = (__bf16)raf[p][1].z; cv.h[7] = (__bf16)raf[p][1].w;
                *reinterpret_cast<uint4*>(&sA[buf][r * LDT + sCol]) = cv.u;
            }
        }
    };

    // One K-stage of WMMAs from buffer `buf`.
    auto compute = [&](int buf) {
#pragma unroll
        for (int kk = 0; kk < BK; kk += 32) {
            FragBF fa[4];
            FragBF fb[2];
            // A frag (16x32 bf16): lane = row M, K runs {laneHi*8..+7} and {+16..+23}
#pragma unroll
            for (int i = 0; i < 4; ++i) {
                const __bf16* p0 =
                    &sA[buf][(waveM * 64 + i * 16 + lane16) * LDT + kk + laneHi * 8];
                fa[i].q[0] = *reinterpret_cast<const uint4*>(p0);
                fa[i].q[1] = *reinterpret_cast<const uint4*>(p0 + 16);
            }
            // B frag (32x16 bf16): lane = col N, K runs {laneHi*8..+7} and {+16..+23}
#pragma unroll
            for (int j = 0; j < 2; ++j) {
                const __bf16* p0 =
                    &sB[buf][(waveN * 32 + j * 16 + lane16) * LDT + kk + laneHi * 8];
                fb[j].q[0] = *reinterpret_cast<const uint4*>(p0);
                fb[j].q[1] = *reinterpret_cast<const uint4*>(p0 + 16);
            }
#pragma unroll
            for (int i = 0; i < 4; ++i)
#pragma unroll
                for (int j = 0; j < 2; ++j)
                    acc[i][j] = __builtin_amdgcn_wmma_f32_16x16x32_bf16(
                        false, fa[i].v, false, fb[j].v,
                        (short)0, acc[i][j], false, false);
        }
    };

    const int S = K / BK;
    issue_async(0, 0);
    load_regsA(0);

    for (int s = 0; s < S; ++s) {
        const int cur = s & 1;
        store_regsA(cur);          // (ATYPE==0) regs for stage s -> sA[cur]
        wait_async0();             // stage-s async copies landed in LDS
        __syncthreads();           // whole tile visible; opposite buffer free
        if (s + 1 < S) {
            issue_async((s + 1) * BK, cur ^ 1);   // overlap DMA with compute
            load_regsA((s + 1) * BK);
        }
        compute(cur);
    }

    // ---------------- epilogue ----------------
    float beta = 0.0f;
    if constexpr (EPI == 1) beta = betaPtr[0];

#pragma unroll
    for (int i = 0; i < 4; ++i) {
#pragma unroll
        for (int j = 0; j < 2; ++j) {
            const int row0 = bm + waveM * 64 + i * 16 + laneHi * 8;
            const int col  = bn + waveN * 32 + j * 16 + lane16;
            const float bb = bias[col];
#pragma unroll
            for (int r = 0; r < 8; ++r) {
                float v = acc[i][j][r] + bb;
                size_t idx = (size_t)(row0 + r) * N + col;
                if constexpr (EPI == 0) {
                    v = fmaxf(v, 0.0f);
                    ((__bf16*)Cout)[idx] = (__bf16)v;
                } else {
                    v += beta * Y[idx];
                    ((float*)Cout)[idx] = v;
                }
            }
        }
    }
}

// ---------------------------------------------------------------------------
// Host-side orchestration
// ---------------------------------------------------------------------------
extern "C" void kernel_launch(void* const* d_in, const int* in_sizes, int n_in,
                              void* d_out, int out_size, void* d_ws, size_t ws_size,
                              hipStream_t stream) {
    const float* x   = (const float*)d_in[0];
    const float* y   = (const float*)d_in[1];
    const float* W1  = (const float*)d_in[2];
    const float* b1  = (const float*)d_in[3];
    const float* W2  = (const float*)d_in[4];
    const float* b2  = (const float*)d_in[5];
    const float* g1  = (const float*)d_in[6];
    const float* bb1 = (const float*)d_in[7];
    const float* g2  = (const float*)d_in[8];
    const float* bb2 = (const float*)d_in[9];
    const float* betaPtr = (const float*)d_in[10];

    const int nfeat  = in_sizes[6];           // 2048
    const int nhid   = in_sizes[8];           // 4096
    const int nclass = in_sizes[5];           // 512
    const int N      = in_sizes[0] / nfeat;   // 32768

    // ---- workspace carve-out (256B aligned) ----
    char* wp = (char*)d_ws;
    auto carve = [&](size_t bytes) -> void* {
        void* p = wp;
        wp += (bytes + 255) & ~(size_t)255;
        return p;
    };
    float* sums1   = (float*)carve((size_t)2 * nfeat * 4);  // sums1 | sumsq1 contiguous
    float* sumsq1  = sums1 + nfeat;
    float* scale1  = (float*)carve((size_t)nfeat * 4);
    float* shift1  = (float*)carve((size_t)nfeat * 4);
    float* b1e     = (float*)carve((size_t)nhid * 4);
    float* sums2   = (float*)carve((size_t)2 * nhid * 4);   // sums2 | sumsq2
    float* sumsq2  = sums2 + nhid;
    float* scale2  = (float*)carve((size_t)nhid * 4);
    float* shift2  = (float*)carve((size_t)nhid * 4);
    float* b2e     = (float*)carve((size_t)nclass * 4);
    void*  W1b     = carve((size_t)nhid * nfeat * 2);       // bf16
    void*  W2b     = carve((size_t)nclass * nhid * 2);      // bf16
    void*  h1b     = carve((size_t)N * nhid * 2);           // bf16 activations

    // ---- Phase 1: BN1 stats over x ----
    {
        int n = 2 * nfeat;
        zero_f32_kernel<<<(n + 255) / 256, 256, 0, stream>>>(sums1, n);
        const int RSPLIT = 64;
        int rpb = (N + RSPLIT - 1) / RSPLIT;
        dim3 g((nfeat + 255) / 256, RSPLIT);
        colsum_f32_kernel<<<g, 256, 0, stream>>>(x, sums1, sumsq1, N, nfeat, rpb);
        bn_finalize_kernel<<<(nfeat + 255) / 256, 256, 0, stream>>>(
            sums1, sumsq1, g1, bb1, scale1, shift1, N, nfeat);
    }

    // ---- Phase 2: fold BN1 into layer 1 ----
    {
        dim3 g((nfeat / 4 + 255) / 256, nhid);
        scale_w_kernel<<<g, 256, 0, stream>>>(W1, scale1, W1b, nfeat);
        fold_bias_kernel<<<nhid, 256, 0, stream>>>(W1, shift1, b1, b1e, nfeat);
    }

    // ---- Phase 3: GEMM1 + ReLU -> h1 (bf16) ----
    {
        dim3 g(nhid / 128, N / 128);   // x = N-blocks (fast) for L2 reuse of A rows
        wmma_gemm_kernel<0, 0><<<g, 256, 0, stream>>>(
            x, W1b, b1e, nullptr, nullptr, h1b, N, nhid, nfeat);
    }

    // ---- Phase 4: BN2 stats over h1 ----
    {
        int n = 2 * nhid;
        zero_f32_kernel<<<(n + 255) / 256, 256, 0, stream>>>(sums2, n);
        const int RSPLIT = 64;
        int rpb = (N + RSPLIT - 1) / RSPLIT;
        dim3 g((nhid + 255) / 256, RSPLIT);
        colsum_bf16_kernel<<<g, 256, 0, stream>>>(h1b, sums2, sumsq2, N, nhid, rpb);
        bn_finalize_kernel<<<(nhid + 255) / 256, 256, 0, stream>>>(
            sums2, sumsq2, g2, bb2, scale2, shift2, N, nhid);
    }

    // ---- Phase 5: fold BN2 into layer 2 ----
    {
        dim3 g((nhid / 4 + 255) / 256, nclass);
        scale_w_kernel<<<g, 256, 0, stream>>>(W2, scale2, W2b, nhid);
        fold_bias_kernel<<<nclass, 256, 0, stream>>>(W2, shift2, b2, b2e, nhid);
    }

    // ---- Phase 6: GEMM2 + b2e + beta*y -> out (fp32) ----
    {
        dim3 g(nclass / 128, N / 128);
        wmma_gemm_kernel<1, 1><<<g, 256, 0, stream>>>(
            h1b, W2b, b2e, y, betaPtr, d_out, N, nclass, nhid);
    }
}
